// TransformerEncoder_80178449481857
// MI455X (gfx1250) — matrix operand order repaired and checked by reading, not compiled, not run on hardware
//
#include <hip/hip_runtime.h>
#include <hip/hip_bf16.h>
#include <math.h>

// Problem constants (from reference): B=4,N=256,D=240,H=8,FF=1024,L=4,HD=30
// Effective sequence: only the "t" half (515 rows) affects the output.
#define BB    4
#define NN    256
#define DD    240
#define DP    256      // D padded to multiple of 32 for WMMA K-dim
#define HH    8
#define HDIM  30       // head dim, padded to 32 in q/k/v buffers
#define FFD   1024
#define LL    4
#define SEQ   515      // 3 pose rows + 2*256 tokens
#define SP    544      // SEQ padded to 34*16 (and 17*32 key chunks)
#define MROWS (BB * SP) // 2176 rows, 136 M-tiles of 16
#define EPSLN 1e-5f

typedef __attribute__((ext_vector_type(16))) _Float16 v16h;
typedef __attribute__((ext_vector_type(8)))  float    v8f;

// ---------------------------------------------------------------------------
// WMMA fragment loaders (wave32, V_WMMA_F32_16X16X32_F16; C/D per ISA 7.12.2, A/B in contiguous k order, not ISA table)
// A (16x32 f16): lane L -> M = L&15; half-group hi = L>>4;
//   element i -> K = hi*16 + i (contiguous, matches B)
//   (per lane this is one contiguous 32B block)
// B (32x16 f16): lane L -> N = L&15; element i -> K = (L>>4)*16 + i (contiguous)
// C/D (16x16 f32): lane L -> N = L&15; acc[r] -> M = r + 8*(L>>4)
// ---------------------------------------------------------------------------
__device__ __forceinline__ v16h load_a16(const _Float16* base, int lda,
                                         int row0, int k0, int lane) {
  int m = lane & 15, hi = lane >> 4;
  const _Float16* rp = base + (size_t)(row0 + m) * lda + k0;
  v16h a;
#pragma unroll
  for (int i = 0; i < 16; ++i) {
    int k = hi * 16 + i;
    a[i] = rp[k];
  }
  return a;
}

// ---------------------------------------------------------------------------
// Generic WMMA GEMM: out[m,n] = act( A[m,:] . W[n,:] + bias[n] (+ resid[m,n]) )
// A: f16 [M][lda], W: f16 [Nout][ldw] (row-major "W" so product is A @ W^T).
// 8 waves/block; each wave computes a 16x32 tile (1 A-frag, 2 B-frags,
// 2 accumulators) so the A fragment is reused across two WMMAs per K-step.
// ---------------------------------------------------------------------------
__global__ __launch_bounds__(256) void gemm_wt_kernel(
    const _Float16* __restrict__ A, int lda,
    const _Float16* __restrict__ W, int ldw,
    const float* __restrict__ bias,
    const float* __restrict__ resid, int ldr,
    float* __restrict__ outF, _Float16* __restrict__ outH, int ldo,
    int Nout, int kIters, int doGelu) {
  int lane = threadIdx.x & 31;
  int wid  = threadIdx.x >> 5;
  int m0 = blockIdx.x * 16;
  int n0 = (blockIdx.y * 8 + wid) * 32;          // 32-wide N chunk per wave
  if (n0 >= Nout) return;                        // wave-uniform
  int hasN1 = (n0 + 16) < Nout;                  // second 16-col sub-tile valid?
  int n = lane & 15, hi = lane >> 4;
  v8f acc0 = {0.f, 0.f, 0.f, 0.f, 0.f, 0.f, 0.f, 0.f};
  v8f acc1 = acc0;
  const _Float16* wrow0 = W + (size_t)(n0 + n) * ldw + hi * 16;
  const _Float16* wrow1 = W + (size_t)(n0 + 16 + (hasN1 ? n : 0)) * ldw + hi * 16;
  for (int kk = 0; kk < kIters; ++kk) {
    int k0 = kk * 32;
    v16h a = load_a16(A, lda, m0, k0, lane);
    v16h b0 = *(const v16h*)(wrow0 + k0);  // 32B aligned: ldw%16==0, k0%32==0
    v16h b1 = *(const v16h*)(wrow1 + k0);
    acc0 = __builtin_amdgcn_wmma_f32_16x16x32_f16(false, a, false, b0,
                                                  (short)0, acc0, false, false);
    acc1 = __builtin_amdgcn_wmma_f32_16x16x32_f16(false, a, false, b1,
                                                  (short)0, acc1, false, false);
  }
  int col0 = n0 + n;
  int col1 = n0 + 16 + n;
  float bc0 = bias ? bias[col0] : 0.f;
  float bc1 = (bias && hasN1) ? bias[col1] : 0.f;
#pragma unroll
  for (int r = 0; r < 8; ++r) {
    int m = m0 + r + hi * 8;
    float v0 = acc0[r] + bc0;
    float v1 = acc1[r] + bc1;
    if (resid) {
      v0 += resid[(size_t)m * ldr + col0];
      if (hasN1) v1 += resid[(size_t)m * ldr + col1];
    }
    if (doGelu) {
      v0 = 0.5f * v0 * (1.f + erff(v0 * 0.70710678118654752f));  // exact gelu
      v1 = 0.5f * v1 * (1.f + erff(v1 * 0.70710678118654752f));
      outH[(size_t)m * ldo + col0] = (_Float16)v0;
      if (hasN1) outH[(size_t)m * ldo + col1] = (_Float16)v1;
    } else {
      outF[(size_t)m * ldo + col0] = v0;
      if (hasN1) outF[(size_t)m * ldo + col1] = v1;
    }
  }
}

// ---------------------------------------------------------------------------
// Assemble the 515 live rows (+ zero padding to 544) into f32/f16 src buffers
// and the coordinate buffer. Also zero the pad columns of attnH once.
// ---------------------------------------------------------------------------
__global__ __launch_bounds__(256) void pack_src_kernel(
    const float* __restrict__ state_t, const float* __restrict__ hand_pose_t,
    const float* __restrict__ head_pose_t, const float* __restrict__ hand_token_t,
    const float* __restrict__ head_token_t,
    const float* __restrict__ coords_hand_t, const float* __restrict__ coords_head_t,
    float* __restrict__ srcF, _Float16* __restrict__ srcH,
    float* __restrict__ cs, _Float16* __restrict__ attnH) {
  int row = blockIdx.x;
  int b = row / SP, s = row % SP;
  int t = threadIdx.x;
  float v = 0.f;
  if (s < SEQ && t < DD) {
    if (s == 0)            v = state_t[b * DD + t];
    else if (s == 1)       v = hand_pose_t[b * DD + t];
    else if (s == 2)       v = head_pose_t[b * DD + t];
    else if (s < 3 + NN)   v = hand_token_t[((size_t)b * NN + (s - 3)) * DD + t];
    else                   v = head_token_t[((size_t)b * NN + (s - 3 - NN)) * DD + t];
  }
  if (t < DD) srcF[(size_t)row * DD + t] = v;
  srcH[(size_t)row * DP + t] = (t < DD) ? (_Float16)v : (_Float16)0.f;
  if (t >= DD) attnH[(size_t)row * DP + t] = (_Float16)0.f;  // pad cols, once
  if (t < 3) {
    float c = 0.f;
    if (s >= 3 && s < 3 + NN)          c = coords_hand_t[((size_t)b * NN + (s - 3)) * 3 + t];
    else if (s >= 3 + NN && s < SEQ)   c = coords_head_t[((size_t)b * NN + (s - 3 - NN)) * 3 + t];
    cs[row * 3 + t] = c;
  }
}

// f32 -> f16 weight conversion with K-dim zero padding.
__global__ __launch_bounds__(256) void cvt_w_kernel(
    const float* __restrict__ W, _Float16* __restrict__ out, int icols, int ocols) {
  int row = blockIdx.x;
  for (int c = threadIdx.x; c < ocols; c += blockDim.x)
    out[(size_t)row * ocols + c] = (c < icols) ? (_Float16)W[(size_t)row * icols + c]
                                               : (_Float16)0.f;
}

// ---------------------------------------------------------------------------
// RoPE (3D, HD=30 -> 3 axes * 5 freq pairs) applied to q,k; pack q,k into
// per-head f16 buffers [B*H][SP][32] (cols 30,31 zero) and V TRANSPOSED into
// vT[B*H][32][SP] so attention's P@V B-fragment is a contiguous 32B load.
// ---------------------------------------------------------------------------
__global__ __launch_bounds__(256) void rope_pack_kernel(
    const float* __restrict__ qf, const float* __restrict__ kf,
    const float* __restrict__ vf, const float* __restrict__ cs,
    _Float16* __restrict__ qh, _Float16* __restrict__ kh, _Float16* __restrict__ vT) {
  int row = blockIdx.x;
  int b = row / SP, s = row % SP;
  int t = threadIdx.x;
  if (t < HH * 15) {  // 8 heads * (3 axes * 5 pairs)
    int h = t / 15, pr = t % 15;
    int a = pr / 5, j = pr % 5;
    float inv = __expf(-0.2f * (float)j * 9.2103403720f);  // 10000^(-2j/10)
    float ang = cs[row * 3 + a] * inv;
    float sn, csn;
    __sincosf(ang, &sn, &csn);
    int c0 = h * HDIM + a * 10 + 2 * j;
    size_t dst = ((size_t)(b * HH + h) * SP + s) * 32 + (a * 10 + 2 * j);
    float q1 = qf[(size_t)row * DD + c0], q2 = qf[(size_t)row * DD + c0 + 1];
    qh[dst]     = (_Float16)(q1 * csn - q2 * sn);
    qh[dst + 1] = (_Float16)(q1 * sn + q2 * csn);
    float k1 = kf[(size_t)row * DD + c0], k2 = kf[(size_t)row * DD + c0 + 1];
    kh[dst]     = (_Float16)(k1 * csn - k2 * sn);
    kh[dst + 1] = (_Float16)(k1 * sn + k2 * csn);
  }
  if (t < DD) {  // V transposed: vT[bh][hd][token]
    int h = t / HDIM, c = t % HDIM;
    vT[((size_t)(b * HH + h) * 32 + c) * SP + s] = (_Float16)vf[(size_t)row * DD + t];
  }
  if (t < HH * 2 * 3) {  // zero pad: q/k cols 30,31 and vT rows 30,31
    int which = t / 16, r2 = t % 16, h = r2 >> 1, c = HDIM + (r2 & 1);
    if (which == 0)      qh[((size_t)(b * HH + h) * SP + s) * 32 + c] = (_Float16)0.f;
    else if (which == 1) kh[((size_t)(b * HH + h) * SP + s) * 32 + c] = (_Float16)0.f;
    else                 vT[((size_t)(b * HH + h) * 32 + c) * SP + s] = (_Float16)0.f;
  }
}

// ---------------------------------------------------------------------------
// Flash attention: one wave per (b,h, 16-query tile). No mask needed within the
// t-half; only padded keys (>=515) are masked. Online softmax; P staged through
// LDS (wave-private tile) to convert C-layout -> A-layout for the P@V WMMAs.
// ---------------------------------------------------------------------------
__global__ __launch_bounds__(256) void attn_kernel(
    const _Float16* __restrict__ qh, const _Float16* __restrict__ kh,
    const _Float16* __restrict__ vT, _Float16* __restrict__ attnH) {
  __shared__ _Float16 pstage[8][16 * 32];  // per-wave 16x32 prob tile
  int lane = threadIdx.x & 31;
  int wid  = threadIdx.x >> 5;
  int qtile = blockIdx.x * 8 + wid;
  int bh = blockIdx.y;
  if (qtile * 16 >= SP) return;  // wave-uniform; EXEC stays all-ones for WMMA
  int b = bh / HH, h = bh % HH;
  const _Float16* Q  = qh + (size_t)bh * SP * 32;
  const _Float16* K  = kh + (size_t)bh * SP * 32;
  const _Float16* Vt = vT + (size_t)bh * 32 * SP;
  int n = lane & 15, hi = lane >> 4;
  int q0 = qtile * 16;
  v16h qfrag = load_a16(Q, 32, q0, 0, lane);
  v8f o0 = {0.f, 0.f, 0.f, 0.f, 0.f, 0.f, 0.f, 0.f};
  v8f o1 = o0;
  v8f zc = o0;
  float rmax[8], rsum[8];
#pragma unroll
  for (int r = 0; r < 8; ++r) { rmax[r] = -1e30f; rsum[r] = 0.f; }
  const float scale = 0.18257418583505536f;  // 1/sqrt(30)
  const _Float16* vrow0 = Vt + (size_t)n * SP + hi * 16;         // hd 0..15
  const _Float16* vrow1 = Vt + (size_t)(16 + n) * SP + hi * 16;  // hd 16..31
  for (int kc = 0; kc < SP / 32; ++kc) {
    int kb = kc * 32;
    // scores = Q(16x32) @ K^T(32x16) for two key groups of 16
    v16h bk0 = *(const v16h*)(K + (size_t)(kb + n) * 32 + hi * 16);
    v16h bk1 = *(const v16h*)(K + (size_t)(kb + 16 + n) * 32 + hi * 16);
    v8f s0 = __builtin_amdgcn_wmma_f32_16x16x32_f16(false, qfrag, false, bk0,
                                                    (short)0, zc, false, false);
    v8f s1 = __builtin_amdgcn_wmma_f32_16x16x32_f16(false, qfrag, false, bk1,
                                                    (short)0, zc, false, false);
    bool mk0 = (kb + n) >= SEQ;
    bool mk1 = (kb + 16 + n) >= SEQ;
#pragma unroll
    for (int r = 0; r < 8; ++r) {
      float a0 = mk0 ? -1e30f : s0[r] * scale;
      float a1 = mk1 ? -1e30f : s1[r] * scale;
      // row reductions stay inside the 16-lane group holding one C-row set
      float cm = fmaxf(a0, a1);
      cm = fmaxf(cm, __shfl_xor(cm, 1, 32));
      cm = fmaxf(cm, __shfl_xor(cm, 2, 32));
      cm = fmaxf(cm, __shfl_xor(cm, 4, 32));
      cm = fmaxf(cm, __shfl_xor(cm, 8, 32));
      float nm = fmaxf(rmax[r], cm);
      float corr = __expf(rmax[r] - nm);
      float p0 = __expf(a0 - nm), p1 = __expf(a1 - nm);
      float ps = p0 + p1;
      ps += __shfl_xor(ps, 1, 32);
      ps += __shfl_xor(ps, 2, 32);
      ps += __shfl_xor(ps, 4, 32);
      ps += __shfl_xor(ps, 8, 32);
      rsum[r] = rsum[r] * corr + ps;
      rmax[r] = nm;
      o0[r] *= corr;
      o1[r] *= corr;
      int m = r + hi * 8;
      pstage[wid][m * 32 + n]      = (_Float16)p0;
      pstage[wid][m * 32 + 16 + n] = (_Float16)p1;
    }
    asm volatile("s_wait_dscnt 0" ::: "memory");  // LDS stores visible to loads
    v16h pf = load_a16(&pstage[wid][0], 32, 0, 0, lane);
    // V chunk B-fragments: contiguous 32B loads from transposed V
    v16h bv0 = *(const v16h*)(vrow0 + kb);
    v16h bv1 = *(const v16h*)(vrow1 + kb);
    o0 = __builtin_amdgcn_wmma_f32_16x16x32_f16(false, pf, false, bv0,
                                                (short)0, o0, false, false);
    o1 = __builtin_amdgcn_wmma_f32_16x16x32_f16(false, pf, false, bv1,
                                                (short)0, o1, false, false);
  }
#pragma unroll
  for (int r = 0; r < 8; ++r) {
    int m = r + hi * 8;
    float inv = 1.f / rsum[r];
    size_t orow = ((size_t)b * SP + q0 + m) * DP;
    attnH[orow + h * HDIM + n] = (_Float16)(o0[r] * inv);
    if (16 + n < HDIM)
      attnH[orow + h * HDIM + 16 + n] = (_Float16)(o1[r] * inv);
  }
}

// LayerNorm over the 240 real columns; emits f32 stream + f16 (padded) copy.
__global__ __launch_bounds__(256) void layernorm_kernel(
    const float* __restrict__ x, const float* __restrict__ g,
    const float* __restrict__ be, float* __restrict__ outF,
    _Float16* __restrict__ outH) {
  __shared__ float red[256];
  int row = blockIdx.x;
  int t = threadIdx.x;
  float v = (t < DD) ? x[(size_t)row * DD + t] : 0.f;
  red[t] = v;
  __syncthreads();
  for (int s = 128; s > 0; s >>= 1) {
    if (t < s) red[t] += red[t + s];
    __syncthreads();
  }
  float mean = red[0] / (float)DD;
  __syncthreads();
  float d = (t < DD) ? (v - mean) : 0.f;
  red[t] = d * d;
  __syncthreads();
  for (int s = 128; s > 0; s >>= 1) {
    if (t < s) red[t] += red[t + s];
    __syncthreads();
  }
  float rstd = rsqrtf(red[0] / (float)DD + EPSLN);
  if (t < DD) {
    float y = (v - mean) * rstd * g[t] + be[t];
    outF[(size_t)row * DD + t] = y;
    outH[(size_t)row * DP + t] = (_Float16)y;
  } else {
    outH[(size_t)row * DP + t] = (_Float16)0.f;
  }
}

__global__ __launch_bounds__(256) void write_out_kernel(
    const float* __restrict__ srcF, float* __restrict__ out) {
  int row = blockIdx.x;  // b*SEQ + q
  int b = row / SEQ, q = row % SEQ;
  int t = threadIdx.x;
  if (t < DD) out[(size_t)row * DD + t] = srcF[((size_t)b * SP + q) * DD + t];
}

// ---------------------------------------------------------------------------
extern "C" void kernel_launch(void* const* d_in, const int* in_sizes, int n_in,
                              void* d_out, int out_size, void* d_ws, size_t ws_size,
                              hipStream_t stream) {
  (void)in_sizes; (void)n_in; (void)out_size; (void)ws_size;
  // setup_inputs() dict order
  const float* hand_token_t  = (const float*)d_in[0];
  const float* head_token_t  = (const float*)d_in[1];
  const float* coords_hand_t = (const float*)d_in[4];
  const float* coords_head_t = (const float*)d_in[5];
  const float* state_t       = (const float*)d_in[8];
  const float* hand_pose_t   = (const float*)d_in[12];
  const float* head_pose_t   = (const float*)d_in[13];
  const float* Wq = (const float*)d_in[14];
  const float* Wk = (const float*)d_in[15];
  const float* Wv = (const float*)d_in[16];
  const float* Wo = (const float*)d_in[17];
  const float* W1 = (const float*)d_in[18];
  const float* W2 = (const float*)d_in[19];
  const float* bq = (const float*)d_in[20];
  const float* bk = (const float*)d_in[21];
  const float* bv = (const float*)d_in[22];
  const float* bo = (const float*)d_in[23];
  const float* b1 = (const float*)d_in[24];
  const float* b2 = (const float*)d_in[25];
  const float* g1 = (const float*)d_in[26];
  const float* be1 = (const float*)d_in[27];
  const float* g2 = (const float*)d_in[28];
  const float* be2 = (const float*)d_in[29];

  char* ws = (char*)d_ws;
  size_t off = 0;
  auto alloc = [&](size_t bytes) -> void* {
    void* p = ws + off;
    off += (bytes + 255) & ~(size_t)255;
    return p;
  };
  float*     srcF  = (float*)alloc((size_t)MROWS * DD * 4);
  _Float16*  srcH  = (_Float16*)alloc((size_t)MROWS * DP * 2);
  float*     cs    = (float*)alloc((size_t)MROWS * 3 * 4);
  _Float16*  wqh   = (_Float16*)alloc((size_t)LL * DD * DP * 2);
  _Float16*  wkh   = (_Float16*)alloc((size_t)LL * DD * DP * 2);
  _Float16*  wvh   = (_Float16*)alloc((size_t)LL * DD * DP * 2);
  _Float16*  woh   = (_Float16*)alloc((size_t)LL * DD * DP * 2);
  _Float16*  w1h   = (_Float16*)alloc((size_t)LL * FFD * DP * 2);
  _Float16*  w2h   = (_Float16*)alloc((size_t)LL * DD * FFD * 2);
  float*     qf    = (float*)alloc((size_t)MROWS * DD * 4);
  float*     kf    = (float*)alloc((size_t)MROWS * DD * 4);
  float*     vf    = (float*)alloc((size_t)MROWS * DD * 4);
  _Float16*  qhB   = (_Float16*)alloc((size_t)BB * HH * SP * 32 * 2);
  _Float16*  khB   = (_Float16*)alloc((size_t)BB * HH * SP * 32 * 2);
  _Float16*  vTB   = (_Float16*)alloc((size_t)BB * HH * 32 * SP * 2);
  _Float16*  attnH = (_Float16*)alloc((size_t)MROWS * DP * 2);
  float*     tmpF  = (float*)alloc((size_t)MROWS * DD * 4);
  float*     s2F   = (float*)alloc((size_t)MROWS * DD * 4);
  _Float16*  s2H   = (_Float16*)alloc((size_t)MROWS * DP * 2);
  _Float16*  ffH   = (_Float16*)alloc((size_t)MROWS * FFD * 2);

  dim3 blk(256);
  pack_src_kernel<<<MROWS, blk, 0, stream>>>(state_t, hand_pose_t, head_pose_t,
                                             hand_token_t, head_token_t,
                                             coords_hand_t, coords_head_t,
                                             srcF, srcH, cs, attnH);
  cvt_w_kernel<<<LL * DD, blk, 0, stream>>>(Wq, wqh, DD, DP);
  cvt_w_kernel<<<LL * DD, blk, 0, stream>>>(Wk, wkh, DD, DP);
  cvt_w_kernel<<<LL * DD, blk, 0, stream>>>(Wv, wvh, DD, DP);
  cvt_w_kernel<<<LL * DD, blk, 0, stream>>>(Wo, woh, DD, DP);
  cvt_w_kernel<<<LL * FFD, blk, 0, stream>>>(W1, w1h, DD, DP);
  cvt_w_kernel<<<LL * DD, blk, 0, stream>>>(W2, w2h, FFD, FFD);

  dim3 gD(MROWS / 16, 1);   // Nout = 240: 8 waves x 32 cols (last sub-tile guarded)
  dim3 gFF(MROWS / 16, 4);  // Nout = 1024: 32 chunks of 32
  dim3 gAttn((SP / 16 + 7) / 8, BB * HH);

  for (int l = 0; l < LL; ++l) {
    const _Float16* wq_l = wqh + (size_t)l * DD * DP;
    const _Float16* wk_l = wkh + (size_t)l * DD * DP;
    const _Float16* wv_l = wvh + (size_t)l * DD * DP;
    const _Float16* wo_l = woh + (size_t)l * DD * DP;
    const _Float16* w1_l = w1h + (size_t)l * FFD * DP;
    const _Float16* w2_l = w2h + (size_t)l * DD * FFD;

    // QKV projections
    gemm_wt_kernel<<<gD, blk, 0, stream>>>(srcH, DP, wq_l, DP, bq + l * DD,
                                           nullptr, 0, qf, nullptr, DD, DD, 8, 0);
    gemm_wt_kernel<<<gD, blk, 0, stream>>>(srcH, DP, wk_l, DP, bk + l * DD,
                                           nullptr, 0, kf, nullptr, DD, DD, 8, 0);
    gemm_wt_kernel<<<gD, blk, 0, stream>>>(srcH, DP, wv_l, DP, bv + l * DD,
                                           nullptr, 0, vf, nullptr, DD, DD, 8, 0);
    rope_pack_kernel<<<MROWS, blk, 0, stream>>>(qf, kf, vf, cs, qhB, khB, vTB);
    attn_kernel<<<gAttn, blk, 0, stream>>>(qhB, khB, vTB, attnH);
    // Wo projection + residual, then LN1
    gemm_wt_kernel<<<gD, blk, 0, stream>>>(attnH, DP, wo_l, DP, bo + l * DD,
                                           srcF, DD, tmpF, nullptr, DD, DD, 8, 0);
    layernorm_kernel<<<MROWS, blk, 0, stream>>>(tmpF, g1 + l * DD, be1 + l * DD,
                                                s2F, s2H);
    // FFN: gelu(s2 @ W1^T + b1) @ W2^T + b2, residual, LN2
    gemm_wt_kernel<<<gFF, blk, 0, stream>>>(s2H, DP, w1_l, DP, b1 + l * FFD,
                                            nullptr, 0, nullptr, ffH, FFD, FFD, 8, 1);
    gemm_wt_kernel<<<gD, blk, 0, stream>>>(ffH, FFD, w2_l, FFD, b2 + l * DD,
                                           s2F, DD, tmpF, nullptr, DD, DD, 32, 0);
    layernorm_kernel<<<MROWS, blk, 0, stream>>>(tmpF, g2 + l * DD, be2 + l * DD,
                                                srcF, srcH);
  }
  write_out_kernel<<<BB * SEQ, blk, 0, stream>>>(srcF, (float*)d_out);
}
